// VishwamaiAttention_48198122995857
// MI455X (gfx1250) — matrix operand order repaired and checked by reading, not compiled, not run on hardware
//
#include <hip/hip_runtime.h>

// ---------------------------------------------------------------------------
// MI455X (gfx1250) attention layer, wave32 + bf16 WMMA (v_wmma_f32_16x16x32_bf16)
// + Tensor Data Mover (tensor_load_to_lds, TENSORcnt) for global->LDS staging.
// B=2, S=2048, HID=4096, H=32, HKV=8, D=128
// Compute-bound (~480 GFLOP vs ~300MB HBM => WMMA bf16, fp32 accumulation)
// ---------------------------------------------------------------------------

#define S_LEN 2048
#define HID   4096
#define NH    32
#define NHKV  8
#define HD    128
#define MROWS 4096            // B*S
#define QKVW  6144            // HID + 2*HKV*D

typedef __attribute__((ext_vector_type(16))) __bf16 v16bf;
typedef __attribute__((ext_vector_type(8)))  float  v8f;
typedef __attribute__((ext_vector_type(4)))  unsigned v4u;
typedef __attribute__((ext_vector_type(8)))  int    v8i;
typedef __attribute__((ext_vector_type(4)))  int    v4i;

#if defined(__has_builtin)
#if __has_builtin(__builtin_amdgcn_tensor_load_to_lds) && \
    __has_builtin(__builtin_amdgcn_s_wait_tensorcnt)
#define HAVE_TDM 1
#endif
#endif
#ifndef HAVE_TDM
#define HAVE_TDM 0
#endif

union Frag16 {
    v16bf bf;
    uint4 u4[2];
};

__device__ __forceinline__ unsigned short f2bf(float f) {
    union { float f; unsigned u; } v; v.f = f;
    unsigned r = v.u + 0x7FFFu + ((v.u >> 16) & 1u);
    return (unsigned short)(r >> 16);
}
__device__ __forceinline__ float bf2f(unsigned short h) {
    union { unsigned u; float f; } v; v.u = ((unsigned)h) << 16;
    return v.f;
}

// Low 32 bits of a generic pointer to LDS == LDS byte address (flat LDS
// aperture maps addr[31:0] to LDS offset).
__device__ __forceinline__ unsigned lds32(const void* p) {
    return (unsigned)(unsigned long long)p;
}

#if HAVE_TDM
// 2D bf16 tile DMA: global (row-major, row_stride elems) -> contiguous LDS.
// D# per CDNA5 ISA ch.8: group0 = count/lds_addr/global_addr/type=2,
// group1 = data_size=1(2B) + tensor dims + tile dims + dim0 stride.
// This toolchain exposes the 6-arg builtin: (g0, g1, g2, g3, g4, cpol).
__device__ __forceinline__ void tdm_load_2d(unsigned lds_byte_addr,
                                            const void* gptr,
                                            unsigned tile_w,   // elems, contig
                                            unsigned tile_h,   // rows
                                            unsigned row_stride_elems)
{
    unsigned long long ga = (unsigned long long)gptr;
    v4u g0;
    g0[0] = 1u;                                      // count=1, user D#
    g0[1] = lds_byte_addr;                           // lds_addr
    g0[2] = (unsigned)(ga & 0xFFFFFFFFu);            // global_addr[31:0]
    g0[3] = (unsigned)((ga >> 32) & 0x01FFFFFFu)     // global_addr[56:32]
          | (2u << 30);                              // type=2 ("image")
    v8i g1;
    g1[0] = (int)(1u << 16);                         // data_size=1 (2 bytes)
    g1[1] = (int)((tile_w & 0xFFFFu) << 16);         // tensor_dim0[15:0]
    g1[2] = (int)((tile_w >> 16) | ((tile_h & 0xFFFFu) << 16)); // td0 hi | td1 lo
    g1[3] = (int)((tile_h >> 16) | (tile_w << 16));  // td1 hi | tile_dim0
    g1[4] = (int)(tile_h & 0xFFFFu);                 // tile_dim1 (tile_dim2=0)
    g1[5] = (int)row_stride_elems;                   // tensor_dim0_stride[31:0]
    g1[6] = 0;
    g1[7] = 0;
    v4i z4 = {0, 0, 0, 0};
    v8i z8 = {0, 0, 0, 0, 0, 0, 0, 0};
    __builtin_amdgcn_tensor_load_to_lds(g0, g1, z4, z4, z8, 0);
}
#endif

// ---------------------------------------------------------------------------
// f32 -> bf16 conversion (vectorized)
// ---------------------------------------------------------------------------
__global__ void cvt4_kernel(const float4* __restrict__ s,
                            ushort4* __restrict__ d, int n4) {
    int i = blockIdx.x * blockDim.x + threadIdx.x;
    if (i >= n4) return;
    float4 v = s[i];
    ushort4 o;
    o.x = f2bf(v.x); o.y = f2bf(v.y); o.z = f2bf(v.z); o.w = f2bf(v.w);
    d[i] = o;
}

// ---------------------------------------------------------------------------
// bf16 GEMM: C[M,N] = A[M,K] @ B[K,N], fp32 accumulation in WMMA.
// Block tile 128x128 (256 thr = 8 waves, 2x4 wave grid, wave tile 64x32).
// Double-buffered LDS; A tile DMA'd by the TDM one K-step ahead (TENSORcnt),
// B tile staged manually (needs transpose). One barrier per K-step.
// ---------------------------------------------------------------------------
template <bool OUT_F32>
__global__ void __launch_bounds__(256)
gemm_bf16(const unsigned short* __restrict__ A, int lda,
          const unsigned short* __restrict__ Bw, int ldb,
          void* __restrict__ Cout, int ldc, int K)
{
    __shared__ __align__(16) unsigned short lA[2][128 * 32];   // row-major
    __shared__ __align__(16) unsigned short lBt[2][128 * 32];  // [n][k]

    const int tid   = threadIdx.x;
    const int lane  = tid & 31;
    const int wave  = tid >> 5;
    const int lanen = lane & 15;
    const int half  = lane >> 4;
    const int wm    = wave >> 2;       // 0..1
    const int wn    = wave & 3;        // 0..3
    const int m0    = blockIdx.y * 128;
    const int n0    = blockIdx.x * 128;

    v8f acc[4][2];
    #pragma unroll
    for (int i = 0; i < 4; i++)
        #pragma unroll
        for (int j = 0; j < 2; j++)
            #pragma unroll
            for (int e = 0; e < 8; e++) acc[i][j][e] = 0.f;

    // ---- prologue: stage tile 0 ----
    {   // B tile 0 (transpose in LDS)
        #pragma unroll
        for (int it = 0; it < 2; it++) {
            int idx = tid + it * 256;
            int kk  = idx >> 4;
            int nc  = (idx & 15) << 3;
            uint4 val = *(const uint4*)(Bw + (size_t)kk * ldb + n0 + nc);
            const unsigned short* e = (const unsigned short*)&val;
            #pragma unroll
            for (int q = 0; q < 8; q++)
                lBt[0][(nc + q) * 32 + kk] = e[q];
        }
    }
#if HAVE_TDM
    if (wave == 0)
        tdm_load_2d(lds32(&lA[0][0]), A + (size_t)m0 * lda, 32, 128,
                    (unsigned)lda);
#else
    {
        int row = tid >> 1;
        int col = (tid & 1) << 4;
        const uint4* gp = (const uint4*)(A + (size_t)(m0 + row) * lda + col);
        uint4* lp = (uint4*)(&lA[0][row * 32 + col]);
        lp[0] = gp[0];
        lp[1] = gp[1];
    }
#endif

    const int T = K >> 5;
    for (int t = 0; t < T; ++t) {
        const int cur = t & 1;
#if HAVE_TDM
        if (wave == 0) __builtin_amdgcn_s_wait_tensorcnt(0);
#endif
        __syncthreads();   // tile t resident in lA[cur]/lBt[cur]

        if (t + 1 < T) {
            const int kn = (t + 1) << 5;
#if HAVE_TDM
            if (wave == 0)
                tdm_load_2d(lds32(&lA[cur ^ 1][0]),
                            A + (size_t)m0 * lda + kn, 32, 128,
                            (unsigned)lda);
#else
            {
                int row = tid >> 1;
                int col = (tid & 1) << 4;
                const uint4* gp =
                    (const uint4*)(A + (size_t)(m0 + row) * lda + kn + col);
                uint4* lp = (uint4*)(&lA[cur ^ 1][row * 32 + col]);
                lp[0] = gp[0];
                lp[1] = gp[1];
            }
#endif
            #pragma unroll
            for (int it = 0; it < 2; it++) {
                int idx = tid + it * 256;
                int kk  = idx >> 4;
                int nc  = (idx & 15) << 3;
                uint4 val =
                    *(const uint4*)(Bw + (size_t)(kn + kk) * ldb + n0 + nc);
                const unsigned short* e = (const unsigned short*)&val;
                #pragma unroll
                for (int q = 0; q < 8; q++)
                    lBt[cur ^ 1][(nc + q) * 32 + kk] = e[q];
            }
            if (t + 2 < T)   // cache prefetch two steps out
                __builtin_prefetch(Bw + (size_t)((t + 2) * 32 + (tid >> 3)) * ldb + n0, 0, 1);
        }

        // ---- fragments + WMMA on tile t ----
        Frag16 afrag[4], bfrag[2];
        #pragma unroll
        for (int mt = 0; mt < 4; mt++) {
            int row = wm * 64 + mt * 16 + lanen;
            const uint4* ap = (const uint4*)(&lA[cur][row * 32]) + half;
            afrag[mt].u4[0] = ap[0];   // K = 8h..8h+7
            afrag[mt].u4[1] = ap[2];   // K = 16+8h..16+8h+7
        }
        #pragma unroll
        for (int nt = 0; nt < 2; nt++) {
            int col = wn * 32 + nt * 16 + lanen;
            const uint4* bp = (const uint4*)(&lBt[cur][col * 32 + half * 8]);
            bfrag[nt].u4[0] = bp[0];   // K = 8h..8h+7
            bfrag[nt].u4[1] = bp[2];   // K = 16+8h..16+8h+7
        }
        #pragma unroll
        for (int mt = 0; mt < 4; mt++)
            #pragma unroll
            for (int nt = 0; nt < 2; nt++)
                acc[mt][nt] = __builtin_amdgcn_wmma_f32_16x16x32_bf16(
                    false, afrag[mt].bf, false, bfrag[nt].bf,
                    (short)0, acc[mt][nt], false, false);
    }

    // ---- epilogue ----
    #pragma unroll
    for (int mt = 0; mt < 4; mt++) {
        #pragma unroll
        for (int nt = 0; nt < 2; nt++) {
            int col = n0 + wn * 32 + nt * 16 + lanen;
            #pragma unroll
            for (int r = 0; r < 8; r++) {
                int row = m0 + wm * 64 + mt * 16 + r + half * 8;
                float v = acc[mt][nt][r];
                if (OUT_F32)
                    ((float*)Cout)[(size_t)row * ldc + col] = v;
                else
                    ((unsigned short*)Cout)[(size_t)row * ldc + col] = f2bf(v);
            }
        }
    }
}

// ---------------------------------------------------------------------------
// RoPE applied in place to the q and k slices of the fused qkv buffer (bf16).
// ---------------------------------------------------------------------------
__global__ void rope_kernel(unsigned short* __restrict__ qkv) {
    int idx = blockIdx.x * blockDim.x + threadIdx.x;   // B*S*(H+HKV)*64 threads
    int pair  = idx & 63;
    int rest  = idx >> 6;
    int head  = rest % (NH + NHKV);
    int rest2 = rest / (NH + NHKV);
    int s = rest2 & (S_LEN - 1);
    int b = rest2 >> 11;
    int colbase = (head < NH) ? head * HD : HID + (head - NH) * HD;
    size_t rowoff = ((size_t)(b * S_LEN + s)) * QKVW + colbase;
    float inv_freq = __powf(10000.0f, -(float)pair * (1.0f / 64.0f));
    float ang = (float)s * inv_freq;
    float sn, cs;
    __sincosf(ang, &sn, &cs);
    float x1 = bf2f(qkv[rowoff + pair]);
    float x2 = bf2f(qkv[rowoff + pair + 64]);
    qkv[rowoff + pair]      = f2bf(x1 * cs - x2 * sn);
    qkv[rowoff + pair + 64] = f2bf(x2 * cs + x1 * sn);
}

// ---------------------------------------------------------------------------
// Flash attention (causal, GQA 4:1). Block = 256 thr (8 waves), one (b,h),
// Q tile of 128 rows (16 per wave, in registers). KV tiles of 64 rows:
// K tile DMA'd by TDM (row-major), V staged manually transposed.
// ---------------------------------------------------------------------------
__global__ void __launch_bounds__(256)
flash_attn(const unsigned short* __restrict__ qkv,
           unsigned short* __restrict__ out)
{
    __shared__ __align__(16) unsigned short lK[64 * 128];    // [krow][d]
    __shared__ __align__(16) unsigned short lVt[128 * 64];   // [d][krow]
    __shared__ __align__(16) unsigned short lP[8][16 * 64];  // per-wave P strip

    const int tid   = threadIdx.x;
    const int lane  = tid & 31;
    const int wave  = tid >> 5;
    const int lanen = lane & 15;
    const int half  = lane >> 4;

    const int q0 = blockIdx.x * 128;
    const int h  = blockIdx.y;
    const int b  = blockIdx.z;
    const int hk = h >> 2;                    // GQA: 4 q-heads per kv-head

    const size_t base = (size_t)b * S_LEN * QKVW;

    // Q fragments: wave's 16 rows x D=128 as 4 K-chunks of 32 (lives in VGPRs)
    Frag16 qfrag[4];
    {
        int qrow = q0 + wave * 16 + lanen;
        const unsigned short* qp = qkv + base + (size_t)qrow * QKVW + h * HD;
        #pragma unroll
        for (int c = 0; c < 4; c++) {
            const uint4* p = (const uint4*)(qp + c * 32) + half;
            qfrag[c].u4[0] = p[0];
            qfrag[c].u4[1] = p[2];
        }
    }

    v8f oacc[8];
    #pragma unroll
    for (int dt = 0; dt < 8; dt++)
        #pragma unroll
        for (int e = 0; e < 8; e++) oacc[dt][e] = 0.f;
    float m_i[8], l_i[8];
    #pragma unroll
    for (int r = 0; r < 8; r++) { m_i[r] = -3.0e38f; l_i[r] = 0.f; }

    const float scale = 0.08838834764831845f;    // 1/sqrt(128)
    const int wave_row_max = q0 + wave * 16 + 15;
    const int nkv = (q0 + 128) >> 6;             // causal KV tile count

    for (int t = 0; t < nkv; t++) {
        const int kv0 = t * 64;
        // --- stage K via TDM, V manually transposed ---
#if HAVE_TDM
        if (wave == 0)
            tdm_load_2d(lds32(lK),
                        qkv + base + (size_t)kv0 * QKVW + HID + hk * HD,
                        128, 64, QKVW);
#endif
        #pragma unroll
        for (int it = 0; it < 4; it++) {
            int idx  = tid + it * 256;         // 0..1023
            int krow = idx >> 4;               // 0..63
            int dc   = (idx & 15) << 3;        // 0..120 step 8
#if !HAVE_TDM
            const unsigned short* kp =
                qkv + base + (size_t)(kv0 + krow) * QKVW + HID + hk * HD + dc;
            *(uint4*)(lK + krow * 128 + dc) = *(const uint4*)kp;
#endif
            const unsigned short* vp =
                qkv + base + (size_t)(kv0 + krow) * QKVW + HID + NHKV * HD + hk * HD + dc;
            uint4 vv = *(const uint4*)vp;
            const unsigned short* e = (const unsigned short*)&vv;
            #pragma unroll
            for (int q = 0; q < 8; q++)
                lVt[(dc + q) * 64 + krow] = e[q];
        }
#if HAVE_TDM
        if (wave == 0) __builtin_amdgcn_s_wait_tensorcnt(0);
#endif
        __syncthreads();

        if (kv0 <= wave_row_max) {
            // --- S = Q @ K^T (16 x 64), K-dim = 128 ---
            v8f sacc[4];
            #pragma unroll
            for (int ct = 0; ct < 4; ct++)
                #pragma unroll
                for (int e = 0; e < 8; e++) sacc[ct][e] = 0.f;
            #pragma unroll
            for (int ct = 0; ct < 4; ct++) {
                int kcol = ct * 16 + lanen;
                #pragma unroll
                for (int c = 0; c < 4; c++) {
                    Frag16 kfrag;
                    const uint4* kp =
                        (const uint4*)(lK + kcol * 128 + c * 32 + half * 8);
                    kfrag.u4[0] = kp[0];
                    kfrag.u4[1] = kp[2];
                    sacc[ct] = __builtin_amdgcn_wmma_f32_16x16x32_bf16(
                        false, qfrag[c].bf, false, kfrag.bf,
                        (short)0, sacc[ct], false, false);
                }
            }
            // --- scale + causal mask ---
            const int rowg0 = q0 + wave * 16 + half * 8;
            #pragma unroll
            for (int ct = 0; ct < 4; ct++) {
                int colg = kv0 + ct * 16 + lanen;
                #pragma unroll
                for (int r = 0; r < 8; r++) {
                    float sv = sacc[ct][r] * scale;
                    if (colg > rowg0 + r) sv = -3.0e38f;
                    sacc[ct][r] = sv;
                }
            }
            // --- online softmax: per-row max/sum across 16-lane half ---
            float corr[8];
            #pragma unroll
            for (int r = 0; r < 8; r++) {
                float v = sacc[0][r];
                #pragma unroll
                for (int ct = 1; ct < 4; ct++) v = fmaxf(v, sacc[ct][r]);
                #pragma unroll
                for (int off = 1; off < 16; off <<= 1)
                    v = fmaxf(v, __shfl_xor(v, off, 32));
                float mnew = fmaxf(m_i[r], v);
                corr[r] = __expf(m_i[r] - mnew);
                m_i[r] = mnew;
            }
            float rsum[8];
            #pragma unroll
            for (int r = 0; r < 8; r++) rsum[r] = 0.f;
            #pragma unroll
            for (int ct = 0; ct < 4; ct++)
                #pragma unroll
                for (int r = 0; r < 8; r++) {
                    float p = __expf(sacc[ct][r] - m_i[r]);
                    sacc[ct][r] = p;
                    rsum[r] += p;
                }
            #pragma unroll
            for (int r = 0; r < 8; r++) {
                float v = rsum[r];
                #pragma unroll
                for (int off = 1; off < 16; off <<= 1)
                    v += __shfl_xor(v, off, 32);
                l_i[r] = l_i[r] * corr[r] + v;
            }
            // --- stage P (bf16) into per-wave LDS region (C-layout -> rowmajor)
            #pragma unroll
            for (int ct = 0; ct < 4; ct++)
                #pragma unroll
                for (int r = 0; r < 8; r++)
                    lP[wave][(r + half * 8) * 64 + ct * 16 + lanen] =
                        f2bf(sacc[ct][r]);
            // --- rescale O, then O += P @ V via WMMA ---
            #pragma unroll
            for (int dt = 0; dt < 8; dt++)
                #pragma unroll
                for (int r = 0; r < 8; r++) oacc[dt][r] *= corr[r];

            Frag16 pfrag[2];
            #pragma unroll
            for (int c = 0; c < 2; c++) {
                const uint4* pp =
                    (const uint4*)(&lP[wave][lanen * 64 + c * 32]) + half;
                pfrag[c].u4[0] = pp[0];
                pfrag[c].u4[1] = pp[2];
            }
            #pragma unroll
            for (int dt = 0; dt < 8; dt++) {
                #pragma unroll
                for (int c = 0; c < 2; c++) {
                    Frag16 vfrag;
                    const uint4* vp =
                        (const uint4*)(lVt + (dt * 16 + lanen) * 64 + c * 32 + half * 8);
                    vfrag.u4[0] = vp[0];
                    vfrag.u4[1] = vp[2];
                    oacc[dt] = __builtin_amdgcn_wmma_f32_16x16x32_bf16(
                        false, pfrag[c].bf, false, vfrag.bf,
                        (short)0, oacc[dt], false, false);
                }
            }
        }
        __syncthreads();
    }

    // --- normalize and write attention output (bf16, [B*S, HID]) ---
    #pragma unroll
    for (int dt = 0; dt < 8; dt++)
        #pragma unroll
        for (int r = 0; r < 8; r++) {
            int row = q0 + wave * 16 + r + half * 8;
            float invl = 1.0f / l_i[r];
            out[(size_t)(b * S_LEN + row) * HID + h * HD + dt * 16 + lanen] =
                f2bf(oacc[dt][r] * invl);
        }
}

// ---------------------------------------------------------------------------
// Launch
// ---------------------------------------------------------------------------
extern "C" void kernel_launch(void* const* d_in, const int* in_sizes, int n_in,
                              void* d_out, int out_size, void* d_ws, size_t ws_size,
                              hipStream_t stream) {
    (void)in_sizes; (void)n_in; (void)out_size; (void)ws_size;
    const float* x  = (const float*)d_in[0];
    const float* wq = (const float*)d_in[1];
    const float* wk = (const float*)d_in[2];
    const float* wv = (const float*)d_in[3];
    const float* wo = (const float*)d_in[4];
    float* out = (float*)d_out;

    // bf16 staging in workspace (~192 MiB)
    unsigned short* xb  = (unsigned short*)d_ws;                    // 4096x4096
    unsigned short* wqb = xb  + (size_t)MROWS * HID;                // 4096x4096
    unsigned short* wkb = wqb + (size_t)HID * HID;                  // 4096x1024
    unsigned short* wvb = wkb + (size_t)HID * (NHKV * HD);          // 4096x1024
    unsigned short* wob = wvb + (size_t)HID * (NHKV * HD);          // 4096x4096
    unsigned short* qkv = wob + (size_t)HID * HID;                  // 4096x6144
    unsigned short* att = qkv + (size_t)MROWS * QKVW;               // 4096x4096

    auto cvt = [&](const float* s, unsigned short* d, size_t n) {
        int n4 = (int)(n >> 2);
        cvt4_kernel<<<(n4 + 255) / 256, 256, 0, stream>>>(
            (const float4*)s, (ushort4*)d, n4);
    };
    cvt(x,  xb,  (size_t)MROWS * HID);
    cvt(wq, wqb, (size_t)HID * HID);
    cvt(wk, wkb, (size_t)HID * (NHKV * HD));
    cvt(wv, wvb, (size_t)HID * (NHKV * HD));
    cvt(wo, wob, (size_t)HID * HID);

    // QKV projection (fused output buffer, row stride 6144)
    gemm_bf16<false><<<dim3(HID / 128, MROWS / 128), 256, 0, stream>>>(
        xb, HID, wqb, HID, qkv, QKVW, HID);
    gemm_bf16<false><<<dim3((NHKV * HD) / 128, MROWS / 128), 256, 0, stream>>>(
        xb, HID, wkb, NHKV * HD, qkv + HID, QKVW, HID);
    gemm_bf16<false><<<dim3((NHKV * HD) / 128, MROWS / 128), 256, 0, stream>>>(
        xb, HID, wvb, NHKV * HD, qkv + HID + NHKV * HD, QKVW, HID);

    // RoPE on q and k slices
    {
        int total = 2 * S_LEN * (NH + NHKV) * 64;
        rope_kernel<<<total / 256, 256, 0, stream>>>(qkv);
    }

    // Causal flash attention
    flash_attn<<<dim3(S_LEN / 128, NH, 2), 256, 0, stream>>>(qkv, att);

    // Output projection -> fp32 result
    gemm_bf16<true><<<dim3(HID / 128, MROWS / 128), 256, 0, stream>>>(
        att, HID, wob, HID, out, HID, HID);
}
